// GAT_26121991095004
// MI455X (gfx1250) — compile-verified, matrix operand order checked by reading, not run on hardware
//
#include <hip/hip_runtime.h>
#include <hip/hip_bf16.h>

typedef __attribute__((ext_vector_type(16))) __bf16 v16bf;
typedef __attribute__((ext_vector_type(8)))  __bf16 v8bf;
typedef __attribute__((ext_vector_type(8)))  float  v8f;

#define N_NODES 10000
#define N_EDGES 160000
#define E_TOT   (N_EDGES + N_NODES)
#define F_IN    128
#define HEADS   8
#define C1      256
#define C2      128
#define C3      64
#define NEG_SLOPE 0.2f

// ---------------- fp32 -> bf16 (round to nearest even) ----------------
__device__ __forceinline__ __bf16 f2bf(float f) {
  unsigned u = __float_as_uint(f);
  unsigned r = (u + 0x7fffu + ((u >> 16) & 1u)) >> 16;
  unsigned short s = (unsigned short)r;
  __bf16 h;
  __builtin_memcpy(&h, &s, 2);
  return h;
}

__global__ void cast_to_bf16(const float* __restrict__ in, __bf16* __restrict__ out, long n) {
  long i = (long)blockIdx.x * blockDim.x + threadIdx.x;
  long stride = (long)gridDim.x * blockDim.x;
  for (; i < n; i += stride) out[i] = f2bf(in[i]);
}

// Weights: cast + transpose  W[K,Nc] -> WT[Nc,K]  so B fragments load contiguously.
__global__ void transpose_cast_bf16(const float* __restrict__ in, __bf16* __restrict__ out,
                                    int K, int Nc) {
  long n = (long)K * Nc;
  long i = (long)blockIdx.x * blockDim.x + threadIdx.x;
  long stride = (long)gridDim.x * blockDim.x;
  for (; i < n; i += stride) {
    int k = (int)(i / Nc), c = (int)(i % Nc);
    out[(long)c * K + k] = f2bf(in[i]);
  }
}

// ---------------- WMMA GEMM: C[M,Nc] = A[M,K] x B[K,Nc] (B given transposed) -----
// 8 waves/block; each wave computes a 16x64 strip = 4 WMMA tiles, reusing A.
// Fragment K pattern per lane: pairs at k = {0,2,4,6,16,18,20,22} + (lane<16?0:8)
// -> two contiguous 8-element (128-bit) runs at ka and ka+16.
__device__ __forceinline__ v16bf load_frag(const __bf16* __restrict__ base, int ka) {
  v8bf lo = *(const v8bf*)(base + ka);
  v8bf hi = *(const v8bf*)(base + ka + 16);
  return __builtin_shufflevector(lo, hi, 0, 1, 2, 3, 4, 5, 6, 7,
                                 8, 9, 10, 11, 12, 13, 14, 15);
}

__global__ __launch_bounds__(256)
void wmma_gemm_bf16(const __bf16* __restrict__ A, const __bf16* __restrict__ BT,
                    float* __restrict__ C, int M, int K, int Nc) {
  int ntn    = Nc >> 6;                         // 64-wide strips
  int nsuper = (M >> 4) * ntn;
  int sup    = blockIdx.x * 8 + (threadIdx.x >> 5);
  if (sup >= nsuper) return;                    // wave-uniform; EXEC stays all-ones
  int lane = threadIdx.x & 31;
  int tm = sup / ntn, tn = sup % ntn;
  int row   = (tm << 4) + (lane & 15);
  int colb  = (tn << 6) + (lane & 15);
  int khalf = (lane < 16) ? 0 : 8;
  const __bf16* arow  = A  + (long)row * K;
  const __bf16* bcol0 = BT + (long)(colb +  0) * K;
  const __bf16* bcol1 = BT + (long)(colb + 16) * K;
  const __bf16* bcol2 = BT + (long)(colb + 32) * K;
  const __bf16* bcol3 = BT + (long)(colb + 48) * K;
  v8f acc0 = {}, acc1 = {}, acc2 = {}, acc3 = {};
  for (int k0 = 0; k0 < K; k0 += 32) {
    int ka = k0 + khalf;
    v16bf a  = load_frag(arow,  ka);
    v16bf b0 = load_frag(bcol0, ka);
    v16bf b1 = load_frag(bcol1, ka);
    v16bf b2 = load_frag(bcol2, ka);
    v16bf b3 = load_frag(bcol3, ka);
    acc0 = __builtin_amdgcn_wmma_f32_16x16x32_bf16(false, a, false, b0, (short)0, acc0, false, false);
    acc1 = __builtin_amdgcn_wmma_f32_16x16x32_bf16(false, a, false, b1, (short)0, acc1, false, false);
    acc2 = __builtin_amdgcn_wmma_f32_16x16x32_bf16(false, a, false, b2, (short)0, acc2, false, false);
    acc3 = __builtin_amdgcn_wmma_f32_16x16x32_bf16(false, a, false, b3, (short)0, acc3, false, false);
  }
  int crow = (tm << 4) + ((lane < 16) ? 0 : 8);
  float* crow0 = C + (long)crow * Nc;
#pragma unroll
  for (int r = 0; r < 8; ++r) {
    float* p = crow0 + (long)r * Nc;
    p[colb +  0] = acc0[r];
    p[colb + 16] = acc1[r];
    p[colb + 32] = acc2[r];
    p[colb + 48] = acc3[r];
  }
}

// ---------------- graph preparation (deterministic CSR) ----------------
__global__ void compute_edges(const int* __restrict__ eidx,
                              int* __restrict__ srcA, int* __restrict__ dstA) {
  int i = blockIdx.x * blockDim.x + threadIdx.x;
  int stride = gridDim.x * blockDim.x;
  for (; i < E_TOT; i += stride) {
    if (i < N_EDGES) { srcA[i] = eidx[i]; dstA[i] = eidx[N_EDGES + i]; }
    else             { srcA[i] = i - N_EDGES; dstA[i] = i - N_EDGES; }
  }
}

__global__ void compute_deg(const int* __restrict__ dstA, int* __restrict__ deg) {
  int i = blockIdx.x * blockDim.x + threadIdx.x;
  int stride = gridDim.x * blockDim.x;
  for (; i < E_TOT; i += stride) atomicAdd(&deg[dstA[i]], 1);
}

__global__ __launch_bounds__(1024)
void scan_rowptr(const int* __restrict__ deg, int* __restrict__ rowptr, int n) {
  __shared__ int part[1024];
  int t = threadIdx.x;
  int chunk = (n + 1023) / 1024;
  int b = t * chunk;
  int s = 0;
  for (int j = 0; j < chunk; ++j) { int i = b + j; if (i < n) s += deg[i]; }
  part[t] = s;
  __syncthreads();
  for (int off = 1; off < 1024; off <<= 1) {    // Hillis-Steele inclusive scan
    int v = (t >= off) ? part[t - off] : 0;
    __syncthreads();
    part[t] += v;
    __syncthreads();
  }
  int run = (t == 0) ? 0 : part[t - 1];         // exclusive prefix for this chunk
  for (int j = 0; j < chunk; ++j) {
    int i = b + j;
    if (i < n) { rowptr[i] = run; run += deg[i]; }
  }
  if (t == 1023) rowptr[n] = part[1023];
}

// Order-preserving fill: each node scans all edges in index order -> deterministic CSR.
__global__ __launch_bounds__(256)
void fill_csr(const int* __restrict__ dstA, const int* __restrict__ rowptr,
              int* __restrict__ csr_eid, int n) {
  int d = blockIdx.x * 256 + threadIdx.x;
  if (d >= n) return;
  int w = rowptr[d];
  for (int i = 0; i < E_TOT; ++i)
    if (dstA[i] == d) csr_eid[w++] = i;
}

// ---------------- GATv2 edge scores: e[eid][h] = att_h . leakyrelu(xl[src]+xr[dst]) ----
__global__ __launch_bounds__(256)
void edge_scores_v2(const float* __restrict__ xl, const float* __restrict__ xr,
                    const int* __restrict__ srcA, const int* __restrict__ dstA,
                    const float* __restrict__ att, float* __restrict__ e, int Cph) {
  int edge = blockIdx.x;
  int t = threadIdx.x;
  int h = t >> 5, lane = t & 31;                // one wave per head
  int s = srcA[edge], d = dstA[edge];
  long HC = (long)HEADS * Cph;
  const float* pl = xl + (long)s * HC + (long)h * Cph;
  const float* pr = xr + (long)d * HC + (long)h * Cph;
  const float* pa = att + h * Cph;
  float sum = 0.f;
  for (int c = lane; c < Cph; c += 32) {
    float v = pl[c] + pr[c];
    v = v > 0.f ? v : NEG_SLOPE * v;
    sum += v * pa[c];
  }
  for (int off = 16; off > 0; off >>= 1) sum += __shfl_down(sum, off, 32);
  if (lane == 0) e[(long)edge * HEADS + h] = sum;
}

// ---------------- per-node softmax + aggregation (GATv2 layers), fused bias+ReLU ----
template<int CPH, int CPT>   // channels/head, channels/thread (CPT = 8*CPH/256)
__global__ __launch_bounds__(256)
void node_agg_v2(const float* __restrict__ xl, const float* __restrict__ e,
                 const int* __restrict__ rowptr, const int* __restrict__ csr_eid,
                 const int* __restrict__ srcA, const float* __restrict__ bias,
                 float* __restrict__ out) {
  int d = blockIdx.x;
  int t = threadIdx.x;
  int h = t >> 5, lane = t & 31;
  int beg = rowptr[d], end = rowptr[d + 1];
  __shared__ float mh[HEADS], dh[HEADS];
  // per-head max (self-loop guarantees >=1 edge)
  float m = -3.0e38f;
  for (int k = beg + lane; k < end; k += 32)
    m = fmaxf(m, e[(long)csr_eid[k] * HEADS + h]);
  for (int off = 16; off > 0; off >>= 1) m = fmaxf(m, __shfl_down(m, off, 32));
  if (lane == 0) mh[h] = m;
  __syncthreads();
  float mm = mh[h];
  // per-head denominator
  float s = 0.f;
  for (int k = beg + lane; k < end; k += 32)
    s += __expf(e[(long)csr_eid[k] * HEADS + h] - mm);
  for (int off = 16; off > 0; off >>= 1) s += __shfl_down(s, off, 32);
  if (lane == 0) dh[h] = s + 1e-16f;
  __syncthreads();
  float inv_d = 1.0f / dh[h];
  // ordered aggregation; thread owns CPT contiguous channels (stays inside its head)
  float acc[CPT];
#pragma unroll
  for (int j = 0; j < CPT; ++j) acc[j] = 0.f;
  int c0 = t * CPT;
  for (int k = beg; k < end; ++k) {
    int eid = csr_eid[k];
    float alpha = __expf(e[(long)eid * HEADS + h] - mm) * inv_d;
    const float* p = xl + (long)srcA[eid] * (HEADS * CPH) + c0;
#pragma unroll
    for (int j = 0; j < CPT; ++j) acc[j] += alpha * p[j];
  }
#pragma unroll
  for (int j = 0; j < CPT; ++j) {
    float v = acc[j] + bias[c0 + j];
    out[(long)d * (HEADS * CPH) + c0 + j] = fmaxf(v, 0.f);   // fused ReLU
  }
}

// ---------------- layer 3 (GATv1, heads=1) ----------------
__global__ void node_att3(const float* __restrict__ xw, const float* __restrict__ att_src,
                          const float* __restrict__ att_dst,
                          float* __restrict__ a_s, float* __restrict__ a_d, int n) {
  int i = blockIdx.x * blockDim.x + threadIdx.x;
  if (i >= n) return;
  const float* p = xw + (long)i * C3;
  float ss = 0.f, sd = 0.f;
  for (int c = 0; c < C3; ++c) { float v = p[c]; ss += v * att_src[c]; sd += v * att_dst[c]; }
  a_s[i] = ss; a_d[i] = sd;
}

__global__ __launch_bounds__(64)
void node_agg_gat(const float* __restrict__ xw, const float* __restrict__ a_s,
                  const float* __restrict__ a_dv,
                  const int* __restrict__ rowptr, const int* __restrict__ csr_eid,
                  const int* __restrict__ srcA, const float* __restrict__ bias,
                  float* __restrict__ out) {
  int d = blockIdx.x, t = threadIdx.x;
  int beg = rowptr[d], end = rowptr[d + 1];
  __shared__ float red[64];
  float ad = a_dv[d];
  float m = -3.0e38f;
  for (int k = beg + t; k < end; k += 64) {
    float v = a_s[srcA[csr_eid[k]]] + ad;
    v = v > 0.f ? v : NEG_SLOPE * v;
    m = fmaxf(m, v);
  }
  red[t] = m; __syncthreads();
  for (int off = 32; off > 0; off >>= 1) {
    if (t < off) red[t] = fmaxf(red[t], red[t + off]);
    __syncthreads();
  }
  float mm = red[0];
  __syncthreads();
  float s = 0.f;
  for (int k = beg + t; k < end; k += 64) {
    float v = a_s[srcA[csr_eid[k]]] + ad;
    v = v > 0.f ? v : NEG_SLOPE * v;
    s += __expf(v - mm);
  }
  red[t] = s; __syncthreads();
  for (int off = 32; off > 0; off >>= 1) {
    if (t < off) red[t] += red[t + off];
    __syncthreads();
  }
  float inv_d = 1.f / (red[0] + 1e-16f);
  float acc = 0.f;
  for (int k = beg; k < end; ++k) {
    int sN = srcA[csr_eid[k]];
    float v = a_s[sN] + ad;
    v = v > 0.f ? v : NEG_SLOPE * v;
    acc += __expf(v - mm) * inv_d * xw[(long)sN * C3 + t];
  }
  out[(long)d * C3 + t] = fmaxf(acc + bias[t], 0.f);   // final ReLU
}

// ---------------- launcher ----------------
extern "C" void kernel_launch(void* const* d_in, const int* in_sizes, int n_in,
                              void* d_out, int out_size, void* d_ws, size_t ws_size,
                              hipStream_t stream) {
  const float* x        = (const float*)d_in[0];
  const int*   eidx     = (const int*)  d_in[1];
  const float* Wl1      = (const float*)d_in[2];
  const float* Wr1      = (const float*)d_in[3];
  const float* att1     = (const float*)d_in[4];
  const float* b1       = (const float*)d_in[5];
  const float* Wl2      = (const float*)d_in[6];
  const float* Wr2      = (const float*)d_in[7];
  const float* att2     = (const float*)d_in[8];
  const float* b2       = (const float*)d_in[9];
  const float* W3       = (const float*)d_in[10];
  const float* att_src3 = (const float*)d_in[11];
  const float* att_dst3 = (const float*)d_in[12];
  const float* b3       = (const float*)d_in[13];

  char* w = (char*)d_ws;
  size_t off = 0;
  auto alloc = [&](size_t bytes) -> void* {
    off = (off + 255) & ~(size_t)255;
    void* p = w + off;
    off += bytes;
    return p;
  };
  const long HC1 = (long)HEADS * C1;   // 2048
  const long HC2 = (long)HEADS * C2;   // 1024
  float*  xl   = (float*) alloc((size_t)N_NODES * HC1 * 4);   // xl1 -> xl2 -> xw3
  float*  xr   = (float*) alloc((size_t)N_NODES * HC1 * 4);   // xr1 -> xr2 -> a_s/a_d
  float*  hbuf = (float*) alloc((size_t)N_NODES * HC1 * 4);   // h1 -> h2
  __bf16* hb   = (__bf16*)alloc((size_t)N_NODES * HC1 * 2);   // bf16 activations (A)
  __bf16* wb   = (__bf16*)alloc((size_t)HC1 * HC2 * 2);       // bf16 transposed weights (B)
  float*  ebuf = (float*) alloc((size_t)E_TOT * HEADS * 4);
  int* srcA    = (int*)   alloc((size_t)E_TOT * 4);
  int* dstA    = (int*)   alloc((size_t)E_TOT * 4);
  int* deg     = (int*)   alloc((size_t)N_NODES * 4);
  int* rowptr  = (int*)   alloc((size_t)(N_NODES + 1) * 4);
  int* csr_eid = (int*)   alloc((size_t)E_TOT * 4);

  // ---- graph prep (deterministic CSR) ----
  compute_edges<<<256, 256, 0, stream>>>(eidx, srcA, dstA);
  hipMemsetAsync(deg, 0, (size_t)N_NODES * 4, stream);
  compute_deg<<<(E_TOT + 255) / 256, 256, 0, stream>>>(dstA, deg);
  scan_rowptr<<<1, 1024, 0, stream>>>(deg, rowptr, N_NODES);
  fill_csr<<<(N_NODES + 255) / 256, 256, 0, stream>>>(dstA, rowptr, csr_eid, N_NODES);

  auto gemm = [&](const __bf16* A, const __bf16* BT, float* C, int M, int K, int Nc) {
    int strips = (M >> 4) * (Nc >> 6);
    wmma_gemm_bf16<<<(strips + 7) / 8, 256, 0, stream>>>(A, BT, C, M, K, Nc);
  };

  // ---- layer 1: GATv2 128 -> 8x256 ----
  cast_to_bf16<<<1024, 256, 0, stream>>>(x, hb, (long)N_NODES * F_IN);
  transpose_cast_bf16<<<1024, 256, 0, stream>>>(Wl1, wb, F_IN, (int)HC1);
  gemm(hb, wb, xl, N_NODES, F_IN, (int)HC1);
  transpose_cast_bf16<<<1024, 256, 0, stream>>>(Wr1, wb, F_IN, (int)HC1);
  gemm(hb, wb, xr, N_NODES, F_IN, (int)HC1);
  edge_scores_v2<<<E_TOT, 256, 0, stream>>>(xl, xr, srcA, dstA, att1, ebuf, C1);
  node_agg_v2<C1, 8><<<N_NODES, 256, 0, stream>>>(xl, ebuf, rowptr, csr_eid, srcA, b1, hbuf);

  // ---- layer 2: GATv2 2048 -> 8x128 ----
  cast_to_bf16<<<1024, 256, 0, stream>>>(hbuf, hb, (long)N_NODES * HC1);
  transpose_cast_bf16<<<1024, 256, 0, stream>>>(Wl2, wb, (int)HC1, (int)HC2);
  gemm(hb, wb, xl, N_NODES, (int)HC1, (int)HC2);
  transpose_cast_bf16<<<1024, 256, 0, stream>>>(Wr2, wb, (int)HC1, (int)HC2);
  gemm(hb, wb, xr, N_NODES, (int)HC1, (int)HC2);
  edge_scores_v2<<<E_TOT, 256, 0, stream>>>(xl, xr, srcA, dstA, att2, ebuf, C2);
  node_agg_v2<C2, 4><<<N_NODES, 256, 0, stream>>>(xl, ebuf, rowptr, csr_eid, srcA, b2, hbuf);

  // ---- layer 3: GATv1 1024 -> 64 ----
  cast_to_bf16<<<1024, 256, 0, stream>>>(hbuf, hb, (long)N_NODES * HC2);
  transpose_cast_bf16<<<1024, 256, 0, stream>>>(W3, wb, (int)HC2, C3);
  float* xw3 = xl;                      // layer-2 xl is dead now
  gemm(hb, wb, xw3, N_NODES, (int)HC2, C3);
  float* a_s = xr;
  float* a_d = xr + N_NODES;
  node_att3<<<(N_NODES + 255) / 256, 256, 0, stream>>>(xw3, att_src3, att_dst3, a_s, a_d, N_NODES);
  node_agg_gat<<<N_NODES, 64, 0, stream>>>(xw3, a_s, a_d, rowptr, csr_eid, srcA, b3,
                                           (float*)d_out);
}
